// SequenceFrameTransformerUpdate_2430951489765
// MI455X (gfx1250) — matrix operand order repaired and checked by reading, not compiled, not run on hardware
//
#include <hip/hip_runtime.h>

typedef __attribute__((ext_vector_type(16))) _Float16 v16h;
typedef __attribute__((ext_vector_type(8)))  _Float16 v8h;
typedef __attribute__((ext_vector_type(8)))  float    v8f;
typedef __attribute__((ext_vector_type(4)))  int      v4i;

namespace cfg {
constexpr int Bc = 4, NRES = 512, NRIG = 2048, CS = 256, CF = 128, CZ = 16;
constexpr int H = 4, CHN = 32, PQ = 8, PV = 12;
constexpr int BQ = 32, BK = 128, NBw = NRIG / BQ;   // 64 windows
constexpr int NBLK = 2;
constexpr int MROW = Bc * NRIG;                     // 8192 rigid rows
constexpr int SROW = Bc * NRES;                     // 2048 residue rows
constexpr int CAT = 336, CATP = 352;                // concat width (padded to 11*32)
constexpr float INFV = 1e8f;
}
using namespace cfg;

#define DEV __device__ __forceinline__

// ---------------------------------------------------------------------------
// Async global->LDS staging (CDNA5 GLOBAL_LOAD_ASYNC_TO_LDS_B128, ASYNCcnt),
// guarded so the build falls back to vectorized copies if the builtin is
// missing on this toolchain.  Toolchain probe showed the builtin's first
// param is a generic int4*, so we pass (v4i*) for global and an
// addrspace(3) int4* for the LDS side.
// ---------------------------------------------------------------------------
#if defined(__gfx1250__) && __has_builtin(__builtin_amdgcn_global_load_async_to_lds_b128)
#define HAS_ASYNC_LDS 1
#else
#define HAS_ASYNC_LDS 0
#endif

typedef __attribute__((address_space(3))) v4i v4i_lds;

DEV void async_copy16(const _Float16* g, _Float16* l) {
#if HAS_ASYNC_LDS
  __builtin_amdgcn_global_load_async_to_lds_b128((v4i*)g, (v4i_lds*)l, 0, 0);
#else
  *(v8h*)l = *(const v8h*)g;
#endif
}
DEV void wait_async_lds() {
#if HAS_ASYNC_LDS
#if __has_builtin(__builtin_amdgcn_s_wait_asynccnt)
  __builtin_amdgcn_s_wait_asynccnt(0);
#else
  asm volatile("s_wait_asynccnt 0x0" ::: "memory");
#endif
#endif
}

// ---------------------------------------------------------------------------
// WMMA helpers (CDNA5 wave32, 16x16x32 f16 -> f32), layouts per ISA §7.12.2.
// All fragment loads are vectorized to 2x ds_load_b128 per lane.
// ---------------------------------------------------------------------------
DEV v8f wmma32(v16h a, v16h b, v8f c) {
  return __builtin_amdgcn_wmma_f32_16x16x32_f16(
      /*neg_a=*/false, a, /*neg_b=*/false, b,
      /*c_mod=*/(short)0, c, /*reuse_a=*/false, /*reuse_b=*/false);
}

// A matrix 16x32 (MxK), row-major, leading dim ld (halves). A lane's 16 halves
// are two contiguous 8-half runs: [k0+hi*8 ..] and [k0+16+hi*8 ..].
DEV v16h load_a_frag(const _Float16* base, int ld, int m0, int k0) {
  const int lane = threadIdx.x & 31;
  const int r = lane & 15, hi = lane >> 4;
  const _Float16* row = base + (size_t)(m0 + r) * ld + k0 + hi * 8;
  v8h lo = *(const v8h*)(row);
  v8h hh = *(const v8h*)(row + 16);
  return __builtin_shufflevector(lo, hh, 0, 1, 2, 3, 4, 5, 6, 7,
                                 8, 9, 10, 11, 12, 13, 14, 15);
}

// B matrix 32x16 (KxN) fragment from an N x K tile (row per output column):
// lane holds column n0+(lane&15), two 8-half runs: [k0+hi*8 ..] and [k0+16+hi*8 ..].
DEV v16h load_bT_frag(const _Float16* base, int ld, int n0, int k0) {
  const int lane = threadIdx.x & 31;
  const int r = lane & 15, hi = lane >> 4;
  const _Float16* row = base + (size_t)(n0 + r) * ld + k0 + hi * 8;
  v8h lo = *(const v8h*)(row);
  v8h hh = *(const v8h*)(row + 16);
  return __builtin_shufflevector(lo, hh, 0, 1, 2, 3, 4, 5, 6, 7,
                                 8, 9, 10, 11, 12, 13, 14, 15);
}

// ---------------------------------------------------------------------------
// Generic f16 WMMA GEMM: C(MxN) = A(MxK) * B(KxN), with B supplied TRANSPOSED
// (Bt is N x K row-major).  Block = 256 thr (8 waves), tile 32(M) x 64(N),
// each wave owns one 16x16 subtile, K-step 32.
// EPI: 0 = f16 store, 1 = f32 store (+bias), 2 = out += v*mask[row],
//      3 = out += gate[row,col]*v*mask[row]
// ---------------------------------------------------------------------------
template <int EPI>
__global__ void __launch_bounds__(256) gemm_f16_kernel(
    const _Float16* __restrict__ A, const _Float16* __restrict__ Bt,
    void* __restrict__ Out, const float* __restrict__ bias,
    const float* __restrict__ gate, const float* __restrict__ mask,
    int M, int N, int K) {
  __shared__ __align__(16) _Float16 sA[32 * 32];   // M x K
  __shared__ __align__(16) _Float16 sB[64 * 32];   // N x K (transposed tile)
  const int m0 = blockIdx.y * 32;
  const int nb = blockIdx.x * 64;
  const int wave = threadIdx.x >> 5;
  const int lane = threadIdx.x & 31;
  const int wm = (wave & 1) * 16;
  const int wn = (wave >> 1) * 16;
  const bool act = (nb + wn) < N;  // wave-uniform
  v8f acc = {0.f, 0.f, 0.f, 0.f, 0.f, 0.f, 0.f, 0.f};
  for (int k0 = 0; k0 < K; k0 += 32) {
    for (int t = threadIdx.x; t < 32 * 4; t += 256) {
      int r = t >> 2, cc = (t & 3) * 8;
      async_copy16(&A[(size_t)(m0 + r) * K + k0 + cc], &sA[r * 32 + cc]);
    }
    if (nb + 64 <= N) {  // full B tile: async DMA path
      for (int t = threadIdx.x; t < 64 * 4; t += 256) {
        int nl = t >> 2, cc = (t & 3) * 8;
        async_copy16(&Bt[(size_t)(nb + nl) * K + k0 + cc], &sB[nl * 32 + cc]);
      }
    } else {             // partial tile: synchronous with zero fill
      for (int t = threadIdx.x; t < 64 * 4; t += 256) {
        int nl = t >> 2, cc = (t & 3) * 8;
        int n = nb + nl;
        v8h v = {};
        if (n < N) v = *(const v8h*)&Bt[(size_t)n * K + k0 + cc];
        *(v8h*)&sB[nl * 32 + cc] = v;
      }
    }
    if (k0 + 32 < K && threadIdx.x < 32) {  // prefetch next A tile
      __builtin_prefetch(&A[(size_t)(m0 + threadIdx.x) * K + k0 + 32], 0, 3);
    }
    wait_async_lds();
    __syncthreads();
    if (act) {
      v16h af = load_a_frag(sA, 32, wm, 0);
      v16h bf = load_bT_frag(sB, 32, wn, 0);
      acc = wmma32(af, bf, acc);
    }
    __syncthreads();
  }
  if (!act) return;
  const int r = lane & 15, hi = lane >> 4;
  const int col = nb + wn + r;
#pragma unroll
  for (int i = 0; i < 8; ++i) {
    const int row = m0 + wm + hi * 8 + i;
    float v = acc[i];
    if (bias) v += bias[col];
    if (EPI == 0) {
      ((_Float16*)Out)[(size_t)row * N + col] = (_Float16)v;
    } else if (EPI == 1) {
      ((float*)Out)[(size_t)row * N + col] = v;
    } else if (EPI == 2) {
      float* o = (float*)Out;
      o[(size_t)row * N + col] += v * mask[row];
    } else {
      float* o = (float*)Out;
      o[(size_t)row * N + col] += gate[(size_t)row * N + col] * v * mask[row];
    }
  }
}

// Dual GEMM for the transition FFN: out = silu(A*W1) * (A*W2), f16 out.
// W1t/W2t are N x K (transposed).
__global__ void __launch_bounds__(256) ffn_kernel(
    const _Float16* __restrict__ A, const _Float16* __restrict__ W1t,
    const _Float16* __restrict__ W2t, _Float16* __restrict__ Out,
    int M, int N, int K) {
  __shared__ __align__(16) _Float16 sA[32 * 32];
  __shared__ __align__(16) _Float16 sB1[64 * 32];
  __shared__ __align__(16) _Float16 sB2[64 * 32];
  const int m0 = blockIdx.y * 32;
  const int nb = blockIdx.x * 64;
  const int wave = threadIdx.x >> 5;
  const int lane = threadIdx.x & 31;
  const int wm = (wave & 1) * 16;
  const int wn = (wave >> 1) * 16;
  v8f acc1 = {0.f, 0.f, 0.f, 0.f, 0.f, 0.f, 0.f, 0.f};
  v8f acc2 = {0.f, 0.f, 0.f, 0.f, 0.f, 0.f, 0.f, 0.f};
  for (int k0 = 0; k0 < K; k0 += 32) {
    for (int t = threadIdx.x; t < 32 * 4; t += 256) {
      int r = t >> 2, cc = (t & 3) * 8;
      async_copy16(&A[(size_t)(m0 + r) * K + k0 + cc], &sA[r * 32 + cc]);
    }
    for (int t = threadIdx.x; t < 64 * 4; t += 256) {
      int nl = t >> 2, cc = (t & 3) * 8;
      int n = nb + nl;
      async_copy16(&W1t[(size_t)n * K + k0 + cc], &sB1[nl * 32 + cc]);
      async_copy16(&W2t[(size_t)n * K + k0 + cc], &sB2[nl * 32 + cc]);
    }
    if (k0 + 32 < K && threadIdx.x < 32) {
      __builtin_prefetch(&A[(size_t)(m0 + threadIdx.x) * K + k0 + 32], 0, 3);
    }
    wait_async_lds();
    __syncthreads();
    v16h af = load_a_frag(sA, 32, wm, 0);
    v16h b1 = load_bT_frag(sB1, 32, wn, 0);
    acc1 = wmma32(af, b1, acc1);
    v16h b2 = load_bT_frag(sB2, 32, wn, 0);
    acc2 = wmma32(af, b2, acc2);
    __syncthreads();
  }
  const int r = lane & 15, hi = lane >> 4;
  const int col = nb + wn + r;
#pragma unroll
  for (int i = 0; i < 8; ++i) {
    const int row = m0 + wm + hi * 8 + i;
    float x = acc1[i], y = acc2[i];
    float si = x / (1.f + __expf(-x));
    Out[(size_t)row * N + col] = (_Float16)(si * y);
  }
}

// ---------------------------------------------------------------------------
// LayerNorm over rows (wave per row), f32 in -> f16 out. w/b optional.
// ---------------------------------------------------------------------------
__global__ void __launch_bounds__(256) ln_rows_kernel(
    const float* __restrict__ X, const float* __restrict__ w,
    const float* __restrict__ b, _Float16* __restrict__ Y, int rows, int C) {
  const int wave = threadIdx.x >> 5, lane = threadIdx.x & 31;
  const int row = blockIdx.x * 8 + wave;
  if (row >= rows) return;
  const float* x = X + (size_t)row * C;
  float s = 0.f, ss = 0.f;
  for (int c = lane; c < C; c += 32) { float v = x[c]; s += v; ss += v * v; }
#pragma unroll
  for (int o = 16; o; o >>= 1) { s += __shfl_xor(s, o, 32); ss += __shfl_xor(ss, o, 32); }
  const float mu = s / C;
  const float inv = rsqrtf(ss / C - mu * mu + 1e-5f);
  for (int c = lane; c < C; c += 32) {
    float v = (x[c] - mu) * inv;
    if (w) v *= w[c];
    if (b) v += b[c];
    Y[(size_t)row * C + c] = (_Float16)v;
  }
}

// f32 -> f16 weight packers, writing TRANSPOSED (N x K) f16 weights.
__global__ void convT_f16(const float* __restrict__ src, _Float16* __restrict__ dst,
                          int K, int N) {
  int i = blockIdx.x * blockDim.x + threadIdx.x;
  if (i >= K * N) return;
  int k = i / N, n = i % N;
  dst[(size_t)n * K + k] = (_Float16)src[i];
}
__global__ void convT_pad_f16(const float* __restrict__ src, _Float16* __restrict__ dst,
                              int K, int N, int Kp) {
  int i = blockIdx.x * blockDim.x + threadIdx.x;
  if (i >= N * Kp) return;
  int n = i / Kp, k = i % Kp;
  dst[i] = (k < K) ? (_Float16)src[(size_t)k * N + n] : (_Float16)0.f;
}
__global__ void conv_f16(const float* __restrict__ src, _Float16* __restrict__ dst, int n) {
  int i = blockIdx.x * blockDim.x + threadIdx.x;
  if (i < n) dst[i] = (_Float16)src[i];
}

// ---------------------------------------------------------------------------
// Rotate/translate the projected points into global frame, compute |pt|^2 sums.
// One thread per (rigid row, head).
// ---------------------------------------------------------------------------
__global__ void __launch_bounds__(256) points_kernel(
    const float* __restrict__ qp, const float* __restrict__ kvp,
    const float* __restrict__ rots, const float* __restrict__ trans,
    _Float16* __restrict__ qpts, _Float16* __restrict__ kpts,
    _Float16* __restrict__ vpts, float* __restrict__ qsq, float* __restrict__ ksq) {
  int t = blockIdx.x * blockDim.x + threadIdx.x;
  if (t >= MROW * H) return;
  const int row = t / H, h = t % H;
  const float* R = rots + (size_t)row * 9;
  const float t0 = trans[(size_t)row * 3 + 0];
  const float t1 = trans[(size_t)row * 3 + 1];
  const float t2 = trans[(size_t)row * 3 + 2];
  float acc = 0.f;
#pragma unroll
  for (int p = 0; p < PQ; ++p) {
    const float* pp = qp + (size_t)row * (H * PQ * 3) + (size_t)(h * PQ + p) * 3;
    float x = pp[0], y = pp[1], z = pp[2];
    float v0 = R[0] * x + R[1] * y + R[2] * z + t0;
    float v1 = R[3] * x + R[4] * y + R[5] * z + t1;
    float v2 = R[6] * x + R[7] * y + R[8] * z + t2;
    _Float16* o = qpts + (size_t)row * (H * 24) + h * 24 + p * 3;
    o[0] = (_Float16)v0; o[1] = (_Float16)v1; o[2] = (_Float16)v2;
    acc += v0 * v0 + v1 * v1 + v2 * v2;
  }
  qsq[row * H + h] = acc;
  acc = 0.f;
#pragma unroll
  for (int p = 0; p < PQ + PV; ++p) {
    const float* pp = kvp + (size_t)row * (H * (PQ + PV) * 3) + (size_t)(h * (PQ + PV) + p) * 3;
    float x = pp[0], y = pp[1], z = pp[2];
    float v0 = R[0] * x + R[1] * y + R[2] * z + t0;
    float v1 = R[3] * x + R[4] * y + R[5] * z + t1;
    float v2 = R[6] * x + R[7] * y + R[8] * z + t2;
    if (p < PQ) {
      _Float16* o = kpts + (size_t)row * (H * 24) + h * 24 + p * 3;
      o[0] = (_Float16)v0; o[1] = (_Float16)v1; o[2] = (_Float16)v2;
      acc += v0 * v0 + v1 * v1 + v2 * v2;
    } else {
      _Float16* o = vpts + (size_t)row * (H * 36) + h * 36 + (p - PQ) * 3;
      o[0] = (_Float16)v0; o[1] = (_Float16)v1; o[2] = (_Float16)v2;
    }
  }
  ksq[row * H + h] = acc;
}

// ---------------------------------------------------------------------------
// IPA windowed attention: one block per (b, window, head). 256 threads.
// ---------------------------------------------------------------------------
__global__ void __launch_bounds__(256) attn_kernel(
    const _Float16* __restrict__ qbuf, const _Float16* __restrict__ kbuf,
    const _Float16* __restrict__ vbuf, const _Float16* __restrict__ qpts,
    const _Float16* __restrict__ kpts, const _Float16* __restrict__ vpts,
    const float* __restrict__ qsq, const float* __restrict__ ksq,
    const float* __restrict__ z, const float* __restrict__ lnzw,
    const float* __restrict__ lnzb, const float* __restrict__ Wb,
    const float* __restrict__ Wdz, const float* __restrict__ head_w,
    const float* __restrict__ mask, const float* __restrict__ rots,
    const float* __restrict__ trans, _Float16* __restrict__ cat) {
  __shared__ __align__(16) _Float16 sQ[BQ * 32];    // q x ch (A-layout)
  __shared__ __align__(16) _Float16 sK[BK * 32];    // key x ch (N x K for Q.K^T)
  __shared__ __align__(16) _Float16 sQP[BQ * 32];   // 24 valid cols, zero padded
  __shared__ __align__(16) _Float16 sKP[BK * 32];
  __shared__ __align__(16) _Float16 sVt[CHN * BK];  // ch x key (N x K for A.V)
  __shared__ __align__(16) _Float16 sVPt[48 * BK];  // ptdim x key (36 valid rows)
  __shared__ __align__(16) _Float16 sPZ[BQ * BK * 4];
  __shared__ __align__(16) float    sL[BQ * BK];    // logits
  __shared__ __align__(16) _Float16 sAtt[BQ * BK];  // softmax(a), A-layout
  __shared__ float    sOPT[BQ * 48];
  __shared__ float    sQSQ[BQ], sKSQ[BK], sQM[BQ], sKM[BK];
  __shared__ int      sKidx[BK];

  const int bid = blockIdx.x;
  const int h = bid & (H - 1);
  const int nb = (bid >> 2) & (NBw - 1);
  const int b = bid >> 8;
  const int tid = threadIdx.x;
  const int wave = tid >> 5, lane = tid & 31;
  const int qrow0 = b * NRIG + nb * BQ;

  // ---- phase 1a: per-key / per-query metadata -----------------------------
  for (int j = tid; j < BK; j += 256) {
    int idx = nb * BQ - 48 + j;  // (BK-BQ)/2 = 48
    bool valid = (idx >= 0) && (idx < NRIG);
    int row = valid ? (b * NRIG + idx) : -1;
    sKidx[j] = row;
    sKM[j] = valid ? mask[row] : 0.f;
    sKSQ[j] = valid ? ksq[row * H + h] : 0.f;
  }
  for (int j = tid; j < BQ; j += 256) {
    sQM[j] = mask[qrow0 + j];
    sQSQ[j] = qsq[(qrow0 + j) * H + h];
  }
  // q-side tiles (async DMA for the full Q tile, sync for padded Qp tile)
  for (int t = tid; t < BQ * 4; t += 256) {
    int q = t >> 2, cc = (t & 3) * 8;
    async_copy16(&qbuf[(size_t)(qrow0 + q) * (H * CHN) + h * CHN + cc],
                 &sQ[q * 32 + cc]);
    v8h vp = {};
    if (cc < 24) vp = *(const v8h*)&qpts[(size_t)(qrow0 + q) * (H * 24) + h * 24 + cc];
    *(v8h*)&sQP[q * 32 + cc] = vp;
  }
  wait_async_lds();
  __syncthreads();
  // ---- phase 1b: gathered key-side tiles ----------------------------------
  for (int t = tid; t < BK * 4; t += 256) {
    int j = t >> 2, cc = (t & 3) * 8;
    int row = sKidx[j];
    v8h vk = {}, vkp = {}, vv = {};
    if (row >= 0) {
      vk = *(const v8h*)&kbuf[(size_t)row * (H * CHN) + h * CHN + cc];
      vv = *(const v8h*)&vbuf[(size_t)row * (H * CHN) + h * CHN + cc];
      if (cc < 24) vkp = *(const v8h*)&kpts[(size_t)row * (H * 24) + h * 24 + cc];
    }
    *(v8h*)&sK[j * 32 + cc] = vk;
    *(v8h*)&sKP[j * 32 + cc] = vkp;
#pragma unroll
    for (int e = 0; e < 8; ++e) sVt[(cc + e) * BK + j] = vv[e];  // transpose V
  }
  for (int t = tid; t < BK * 48; t += 256) {
    int j = t / 48, c = t % 48;
    int row = sKidx[j];
    sVPt[c * BK + j] =
        (row >= 0 && c < 36) ? vpts[(size_t)row * (H * 36) + h * 36 + c] : (_Float16)0.f;
  }
  // ---- phase 2: z LayerNorm -> bbias / pair_z, init logits ----------------
  const float whv = logf(1.f + __expf(head_w[h])) * 0.09622504486493763f; // sqrt(1/108)
  {
    const float4* z4 = (const float4*)(z + (size_t)((b * NBw + nb) * BQ) * BK * CZ);
    for (int t = tid; t < BQ * BK; t += 256) {
      int q = t >> 7, k = t & 127;
      float4 a0 = z4[(size_t)t * 4 + 0];
      float4 a1 = z4[(size_t)t * 4 + 1];
      float4 a2 = z4[(size_t)t * 4 + 2];
      float4 a3 = z4[(size_t)t * 4 + 3];
      float zz[16] = {a0.x, a0.y, a0.z, a0.w, a1.x, a1.y, a1.z, a1.w,
                      a2.x, a2.y, a2.z, a2.w, a3.x, a3.y, a3.z, a3.w};
      float s = 0.f, ssq = 0.f;
#pragma unroll
      for (int c = 0; c < 16; ++c) { s += zz[c]; ssq += zz[c] * zz[c]; }
      float mu = s * (1.f / 16.f);
      float inv = rsqrtf(ssq * (1.f / 16.f) - mu * mu + 1e-5f);
      float bb = 0.f, pz0 = 0.f, pz1 = 0.f, pz2 = 0.f, pz3 = 0.f;
#pragma unroll
      for (int c = 0; c < 16; ++c) {
        float y = (zz[c] - mu) * inv * lnzw[c] + lnzb[c];
        bb += y * Wb[c * H + h];
        pz0 += y * Wdz[c * 4 + 0];
        pz1 += y * Wdz[c * 4 + 1];
        pz2 += y * Wdz[c * 4 + 2];
        pz3 += y * Wdz[c * 4 + 3];
      }
      sPZ[t * 4 + 0] = (_Float16)pz0;
      sPZ[t * 4 + 1] = (_Float16)pz1;
      sPZ[t * 4 + 2] = (_Float16)pz2;
      sPZ[t * 4 + 3] = (_Float16)pz3;
      float mb = (sQM[q] * sKM[k] - 1.f) * INFV;
      sL[t] = 0.5773502691896258f * bb - 0.5f * whv * (sQSQ[q] + sKSQ[k]) + mb;
    }
  }
  __syncthreads();
  // ---- phase 3: logits += scale*Q.K^T + wh*Qp.Kp^T (WMMA) -----------------
  for (int tt = wave; tt < 16; tt += 8) {
    const int m0 = (tt & 1) * 16;
    const int n0 = (tt >> 1) * 16;
    v8f s1 = {0.f, 0.f, 0.f, 0.f, 0.f, 0.f, 0.f, 0.f};
    v8f s2 = {0.f, 0.f, 0.f, 0.f, 0.f, 0.f, 0.f, 0.f};
    v16h aq = load_a_frag(sQ, 32, m0, 0);
    v16h bk = load_bT_frag(sK, 32, n0, 0);
    s1 = wmma32(aq, bk, s1);
    v16h ap = load_a_frag(sQP, 32, m0, 0);
    v16h bp = load_bT_frag(sKP, 32, n0, 0);
    s2 = wmma32(ap, bp, s2);
    const int r = lane & 15, hi = lane >> 4;
#pragma unroll
    for (int i = 0; i < 8; ++i) {
      int q = m0 + hi * 8 + i, k = n0 + r;
      sL[q * BK + k] += 0.10206207261596577f * s1[i] + whv * s2[i];
    }
  }
  __syncthreads();
  // ---- phase 4: softmax over keys -----------------------------------------
  if (tid < BQ) {
    const int q = tid;
    float mx = -3.0e38f;
    for (int k = 0; k < BK; ++k) mx = fmaxf(mx, sL[q * BK + k]);
    float s = 0.f;
    for (int k = 0; k < BK; ++k) {
      float e = __expf(sL[q * BK + k] - mx);
      sL[q * BK + k] = e;
      s += e;
    }
    float invs = 1.f / (s + 1e-20f);
    for (int k = 0; k < BK; ++k) sAtt[q * BK + k] = (_Float16)(sL[q * BK + k] * invs);
  }
  __syncthreads();
  // ---- phase 5: O = A.V, OPT = A.Vpts (WMMA) ------------------------------
  for (int tt = wave; tt < 10; tt += 8) {
    const int r = lane & 15, hi = lane >> 4;
    if (tt < 4) {
      const int m0 = (tt & 1) * 16, n0 = (tt >> 1) * 16;
      v8f acc = {0.f, 0.f, 0.f, 0.f, 0.f, 0.f, 0.f, 0.f};
#pragma unroll
      for (int k0 = 0; k0 < BK; k0 += 32) {
        v16h af = load_a_frag(sAtt, BK, m0, k0);
        v16h bf = load_bT_frag(sVt, BK, n0, k0);
        acc = wmma32(af, bf, acc);
      }
#pragma unroll
      for (int i = 0; i < 8; ++i) {
        int q = m0 + hi * 8 + i, n = n0 + r;
        cat[(size_t)(qrow0 + q) * CATP + h * CHN + n] = (_Float16)acc[i];
      }
    } else {
      const int u = tt - 4;
      const int m0 = (u & 1) * 16, n0 = (u >> 1) * 16;
      v8f acc = {0.f, 0.f, 0.f, 0.f, 0.f, 0.f, 0.f, 0.f};
#pragma unroll
      for (int k0 = 0; k0 < BK; k0 += 32) {
        v16h af = load_a_frag(sAtt, BK, m0, k0);
        v16h bf = load_bT_frag(sVPt, BK, n0, k0);
        acc = wmma32(af, bf, acc);
      }
#pragma unroll
      for (int i = 0; i < 8; ++i) {
        int q = m0 + hi * 8 + i, n = n0 + r;
        sOPT[q * 48 + n] = acc[i];
      }
    }
  }
  __syncthreads();
  // ---- phase 6: inverse-frame transform of o_pt + norms -------------------
  for (int t = tid; t < BQ * PV; t += 256) {
    const int q = t / PV, vv = t % PV;
    const int grow = qrow0 + q;
    float d0 = sOPT[q * 48 + vv * 3 + 0] - trans[(size_t)grow * 3 + 0];
    float d1 = sOPT[q * 48 + vv * 3 + 1] - trans[(size_t)grow * 3 + 1];
    float d2 = sOPT[q * 48 + vv * 3 + 2] - trans[(size_t)grow * 3 + 2];
    const float* R = rots + (size_t)grow * 9;
    float r0 = R[0] * d0 + R[3] * d1 + R[6] * d2;  // R^T
    float r1 = R[1] * d0 + R[4] * d1 + R[7] * d2;
    float r2 = R[2] * d0 + R[5] * d1 + R[8] * d2;
    float nm = sqrtf(r0 * r0 + r1 * r1 + r2 * r2 + 1e-8f);
    _Float16* crow = cat + (size_t)grow * CATP;
    crow[128 + h * 36 + vv * 3 + 0] = (_Float16)r0;
    crow[128 + h * 36 + vv * 3 + 1] = (_Float16)r1;
    crow[128 + h * 36 + vv * 3 + 2] = (_Float16)r2;
    crow[272 + h * PV + vv] = (_Float16)nm;
  }
  // ---- phase 7: o_pair -----------------------------------------------------
  for (int t = tid; t < BQ * 4; t += 256) {
    const int q = t >> 2, c = t & 3;
    float s = 0.f;
    for (int k = 0; k < BK; ++k)
      s += (float)sAtt[q * BK + k] * (float)sPZ[(q * BK + k) * 4 + c];
    cat[(size_t)(qrow0 + q) * CATP + 320 + h * 4 + c] = (_Float16)s;
  }
  // ---- phase 8: zero the K-padding columns --------------------------------
  if (h == 0) {
    for (int t = tid; t < BQ * 16; t += 256) {
      int q = t >> 4, c = t & 15;
      cat[(size_t)(qrow0 + q) * CATP + CAT + c] = (_Float16)0.f;
    }
  }
}

// ---------------------------------------------------------------------------
// Transition s2: LN(re) * sigmoid(gather(Acg)) + gather(Acb); cg = sigmoid(Ag)
// ---------------------------------------------------------------------------
__global__ void __launch_bounds__(256) s2_kernel(
    const float* __restrict__ re, const int* __restrict__ idxmap,
    const float* __restrict__ Acg, const float* __restrict__ Acb,
    const float* __restrict__ Ag, _Float16* __restrict__ s2,
    float* __restrict__ cg, int rows) {
  const int wave = threadIdx.x >> 5, lane = threadIdx.x & 31;
  const int row = blockIdx.x * 8 + wave;
  if (row >= rows) return;
  const int b = row / NRIG;
  const int g = b * NRES + idxmap[row];
  const float* x = re + (size_t)row * CF;
  float s = 0.f, ss = 0.f;
  for (int c = lane; c < CF; c += 32) { float v = x[c]; s += v; ss += v * v; }
#pragma unroll
  for (int o = 16; o; o >>= 1) { s += __shfl_xor(s, o, 32); ss += __shfl_xor(ss, o, 32); }
  const float mu = s / CF;
  const float inv = rsqrtf(ss / CF - mu * mu + 1e-5f);
  for (int c = lane; c < CF; c += 32) {
    float sln = (x[c] - mu) * inv;
    float a1 = Acg[(size_t)g * CF + c];
    float a2 = Acb[(size_t)g * CF + c];
    float a3 = Ag[(size_t)g * CF + c];
    float sg = 1.f / (1.f + __expf(-a1));
    s2[(size_t)row * CF + c] = (_Float16)(sln * sg + a2);
    cg[(size_t)row * CF + c] = 1.f / (1.f + __expf(-a3));
  }
}

// ---------------------------------------------------------------------------
// Host orchestration
// ---------------------------------------------------------------------------
extern "C" void kernel_launch(void* const* d_in, const int* in_sizes, int n_in,
                              void* d_out, int out_size, void* d_ws, size_t ws_size,
                              hipStream_t stream) {
  const float* s_in  = (const float*)d_in[0];
  const float* re_in = (const float*)d_in[1];
  const float* rots  = (const float*)d_in[2];
  const float* trans = (const float*)d_in[3];
  const float* zfp   = (const float*)d_in[4];
  const float* maskp = (const float*)d_in[5];
  const int*   idxp  = (const int*)d_in[6];
  const float* ln_s_w = (const float*)d_in[7];
  const float* ln_s_b = (const float*)d_in[8];
  const float* ln_z_w = (const float*)d_in[9];
  const float* ln_z_b = (const float*)d_in[10];
  const float* Wq   = (const float*)d_in[11];
  const float* Wk   = (const float*)d_in[12];
  const float* Wv   = (const float*)d_in[13];
  const float* Wqp  = (const float*)d_in[14];
  const float* Wkvp = (const float*)d_in[15];
  const float* Wbp  = (const float*)d_in[16];
  const float* Wdz  = (const float*)d_in[17];
  const float* hwp  = (const float*)d_in[18];
  const float* Wout = (const float*)d_in[19];
  const float* lcw  = (const float*)d_in[20];
  const float* Wcg  = (const float*)d_in[21];
  const float* bcg  = (const float*)d_in[22];
  const float* Wcb  = (const float*)d_in[23];
  const float* W1   = (const float*)d_in[24];
  const float* W2   = (const float*)d_in[25];
  const float* Wb2  = (const float*)d_in[26];
  const float* Wg   = (const float*)d_in[27];
  const float* bg   = (const float*)d_in[28];
  float* out = (float*)d_out;

  char* base = (char*)d_ws;
  size_t off = 0;
  auto alloc = [&](size_t bytes) -> char* {
    char* p = base + off;
    off = (off + bytes + 255) & ~(size_t)255;
    return p;
  };
  _Float16* se_ln = (_Float16*)alloc((size_t)MROW * CF * 2);
  _Float16* qbuf  = (_Float16*)alloc((size_t)MROW * 128 * 2);
  _Float16* kbuf  = (_Float16*)alloc((size_t)MROW * 128 * 2);
  _Float16* vbuf  = (_Float16*)alloc((size_t)MROW * 128 * 2);
  float*    qp    = (float*)alloc((size_t)MROW * 96 * 4);
  float*    kvp   = (float*)alloc((size_t)MROW * 240 * 4);
  _Float16* qpts  = (_Float16*)alloc((size_t)MROW * 96 * 2);
  _Float16* kpts  = (_Float16*)alloc((size_t)MROW * 96 * 2);
  _Float16* vpts  = (_Float16*)alloc((size_t)MROW * 144 * 2);
  float*    qsq   = (float*)alloc((size_t)MROW * H * 4);
  float*    ksq   = (float*)alloc((size_t)MROW * H * 4);
  _Float16* cat   = (_Float16*)alloc((size_t)MROW * CATP * 2);
  _Float16* sf16  = (_Float16*)alloc((size_t)SROW * CS * 2);
  _Float16* cln   = (_Float16*)alloc((size_t)SROW * CS * 2);
  float*    Acg   = (float*)alloc((size_t)SROW * CF * 4);
  float*    Acb   = (float*)alloc((size_t)SROW * CF * 4);
  float*    Ag    = (float*)alloc((size_t)SROW * CF * 4);
  _Float16* s2b   = (_Float16*)alloc((size_t)MROW * CF * 2);
  float*    cgb   = (float*)alloc((size_t)MROW * CF * 4);
  _Float16* bpart = (_Float16*)alloc((size_t)MROW * 2 * CF * 2);
  _Float16* wq_h   = (_Float16*)alloc((size_t)CF * 128 * 2);
  _Float16* wk_h   = (_Float16*)alloc((size_t)CF * 128 * 2);
  _Float16* wv_h   = (_Float16*)alloc((size_t)CF * 128 * 2);
  _Float16* wqp_h  = (_Float16*)alloc((size_t)CF * 96 * 2);
  _Float16* wkvp_h = (_Float16*)alloc((size_t)CF * 240 * 2);
  _Float16* wout_h = (_Float16*)alloc((size_t)CATP * CF * 2);
  _Float16* wcg_h  = (_Float16*)alloc((size_t)CS * CF * 2);
  _Float16* wcb_h  = (_Float16*)alloc((size_t)CS * CF * 2);
  _Float16* wg_h   = (_Float16*)alloc((size_t)CS * CF * 2);
  _Float16* w1_h   = (_Float16*)alloc((size_t)CF * 2 * CF * 2);
  _Float16* w2_h   = (_Float16*)alloc((size_t)CF * 2 * CF * 2);
  _Float16* wb2_h  = (_Float16*)alloc((size_t)2 * CF * CF * 2);
  (void)ws_size; (void)in_sizes; (void)n_in; (void)out_size;

  auto cgrid = [](int n) { return dim3((unsigned)((n + 255) / 256)); };

  // running residual lives in d_out
  (void)hipMemcpyAsync(d_out, re_in, (size_t)MROW * CF * 4, hipMemcpyDeviceToDevice, stream);
  conv_f16<<<cgrid(SROW * CS), 256, 0, stream>>>(s_in, sf16, SROW * CS);

  for (int i = 0; i < NBLK; ++i) {
    // ---- weight packing: f16, transposed to N x K ----
    convT_f16<<<cgrid(CF * 128), 256, 0, stream>>>(Wq + (size_t)i * CF * 128, wq_h, CF, 128);
    convT_f16<<<cgrid(CF * 128), 256, 0, stream>>>(Wk + (size_t)i * CF * 128, wk_h, CF, 128);
    convT_f16<<<cgrid(CF * 128), 256, 0, stream>>>(Wv + (size_t)i * CF * 128, wv_h, CF, 128);
    convT_f16<<<cgrid(CF * 96), 256, 0, stream>>>(Wqp + (size_t)i * CF * 96, wqp_h, CF, 96);
    convT_f16<<<cgrid(CF * 240), 256, 0, stream>>>(Wkvp + (size_t)i * CF * 240, wkvp_h, CF, 240);
    convT_pad_f16<<<cgrid(CF * CATP), 256, 0, stream>>>(Wout + (size_t)i * CAT * CF, wout_h, CAT, CF, CATP);
    convT_f16<<<cgrid(CS * CF), 256, 0, stream>>>(Wcg + (size_t)i * CS * CF, wcg_h, CS, CF);
    convT_f16<<<cgrid(CS * CF), 256, 0, stream>>>(Wcb + (size_t)i * CS * CF, wcb_h, CS, CF);
    convT_f16<<<cgrid(CS * CF), 256, 0, stream>>>(Wg + (size_t)i * CS * CF, wg_h, CS, CF);
    convT_f16<<<cgrid(CF * 2 * CF), 256, 0, stream>>>(W1 + (size_t)i * CF * 2 * CF, w1_h, CF, 2 * CF);
    convT_f16<<<cgrid(CF * 2 * CF), 256, 0, stream>>>(W2 + (size_t)i * CF * 2 * CF, w2_h, CF, 2 * CF);
    convT_f16<<<cgrid(2 * CF * CF), 256, 0, stream>>>(Wb2 + (size_t)i * 2 * CF * CF, wb2_h, 2 * CF, CF);

    // ---- IPA ----
    ln_rows_kernel<<<dim3(MROW / 8), 256, 0, stream>>>(out, ln_s_w + i * CF, ln_s_b + i * CF, se_ln, MROW, CF);
    gemm_f16_kernel<0><<<dim3(2, MROW / 32), 256, 0, stream>>>(se_ln, wq_h, qbuf, nullptr, nullptr, nullptr, MROW, 128, CF);
    gemm_f16_kernel<0><<<dim3(2, MROW / 32), 256, 0, stream>>>(se_ln, wk_h, kbuf, nullptr, nullptr, nullptr, MROW, 128, CF);
    gemm_f16_kernel<0><<<dim3(2, MROW / 32), 256, 0, stream>>>(se_ln, wv_h, vbuf, nullptr, nullptr, nullptr, MROW, 128, CF);
    gemm_f16_kernel<1><<<dim3(2, MROW / 32), 256, 0, stream>>>(se_ln, wqp_h, qp, nullptr, nullptr, nullptr, MROW, 96, CF);
    gemm_f16_kernel<1><<<dim3(4, MROW / 32), 256, 0, stream>>>(se_ln, wkvp_h, kvp, nullptr, nullptr, nullptr, MROW, 240, CF);
    points_kernel<<<cgrid(MROW * H), 256, 0, stream>>>(qp, kvp, rots, trans, qpts, kpts, vpts, qsq, ksq);
    attn_kernel<<<dim3(Bc * NBw * H), 256, 0, stream>>>(
        qbuf, kbuf, vbuf, qpts, kpts, vpts, qsq, ksq, zfp,
        ln_z_w + i * CZ, ln_z_b + i * CZ, Wbp + (size_t)i * CZ * H,
        Wdz + (size_t)i * CZ * 4, hwp + i * H, maskp, rots, trans, cat);
    gemm_f16_kernel<2><<<dim3(2, MROW / 32), 256, 0, stream>>>(cat, wout_h, out, nullptr, nullptr, maskp, MROW, CF, CATP);

    // ---- transition ----
    ln_rows_kernel<<<dim3(SROW / 8), 256, 0, stream>>>(s_in, lcw + i * CS, nullptr, cln, SROW, CS);
    gemm_f16_kernel<1><<<dim3(2, SROW / 32), 256, 0, stream>>>(cln, wcg_h, Acg, bcg + i * CF, nullptr, nullptr, SROW, CF, CS);
    gemm_f16_kernel<1><<<dim3(2, SROW / 32), 256, 0, stream>>>(cln, wcb_h, Acb, nullptr, nullptr, nullptr, SROW, CF, CS);
    gemm_f16_kernel<1><<<dim3(2, SROW / 32), 256, 0, stream>>>(sf16, wg_h, Ag, bg + i * CF, nullptr, nullptr, SROW, CF, CS);
    s2_kernel<<<dim3(MROW / 8), 256, 0, stream>>>(out, idxp, Acg, Acb, Ag, s2b, cgb, MROW);
    ffn_kernel<<<dim3(4, MROW / 32), 256, 0, stream>>>(s2b, w1_h, w2_h, bpart, MROW, 2 * CF, CF);
    gemm_f16_kernel<3><<<dim3(2, MROW / 32), 256, 0, stream>>>(bpart, wb2_h, out, nullptr, cgb, maskp, MROW, CF, 2 * CF);
  }
}
